// DitTalkingHead_21474836480607
// MI455X (gfx1250) — matrix operand order repaired and checked by reading, not compiled, not run on hardware
//
#include <hip/hip_runtime.h>
#include <hip/hip_bf16.h>

typedef _Float16 half8   __attribute__((ext_vector_type(8)));
typedef _Float16 half16  __attribute__((ext_vector_type(16)));
typedef float    float8  __attribute__((ext_vector_type(8)));
typedef unsigned int uint4v __attribute__((ext_vector_type(4)));
typedef int      int4v   __attribute__((ext_vector_type(4)));
typedef int      int8v   __attribute__((ext_vector_type(8)));

constexpr int B_  = 4;
constexpr int L_  = 2048;
constexpr int D_  = 512;
constexpr int H_  = 8;
constexpr int HD_ = 64;
constexpr int BHL_ = B_ * H_ * L_;   // 65536

#if defined(__has_builtin)
#  if __has_builtin(__builtin_amdgcn_tensor_load_to_lds)
#    define HAVE_TDM 1
#  else
#    define HAVE_TDM 0
#  endif
#else
#  define HAVE_TDM 0
#endif

// ---------- helpers ----------

__device__ __forceinline__ half16 ld2x8(const _Float16* plo, const _Float16* phi) {
  half8 lo = *reinterpret_cast<const half8*>(plo);
  half8 hi = *reinterpret_cast<const half8*>(phi);
  half16 r;
#pragma unroll
  for (int i = 0; i < 8; ++i) { r[i] = lo[i]; r[i + 8] = hi[i]; }
  return r;
}

__device__ __forceinline__ float8 wmma_f16(half16 a, half16 b, float8 c) {
  // D = A(16x32 f16) * B(32x16 f16) + C(16x16 f32)
  return __builtin_amdgcn_wmma_f32_16x16x32_f16(false, a, false, b, (short)0, c,
                                                false, false);
}

#if HAVE_TDM
// TDM: stage one contiguous 4KB tile (512 x 8B elements) from global into LDS.
// D# per CDNA5 ISA (1-D tile, data_size=8B).
__device__ __forceinline__ void tdm_load_tile_4k(const void* gsrc, void* lds_dst) {
  unsigned long long ga = (unsigned long long)(uintptr_t)gsrc;
  unsigned int       la = (unsigned int)(uintptr_t)lds_dst;  // flat addr[31:0] == LDS addr

  uint4v g0 = (uint4v)0u;
  g0[0] = 1u;                                   // count=1 (valid descriptor), user mode
  g0[1] = la;                                   // lds_addr (bytes)
  g0[2] = (unsigned int)ga;                     // global_addr[31:0]
  g0[3] = (unsigned int)((ga >> 32) & 0x01FFFFFFu) | (2u << 30);  // ga[56:32] | type=2

  const unsigned int td0 = 512u;                // tensor_dim0 (8B units)
  const unsigned int tl0 = 512u;                // tile_dim0
  int8v g1 = (int8v)0;
  g1[0] = (int)(3u << 16);                      // wg_mask=0, data_size=3 (8 bytes)
  g1[1] = (int)((td0 & 0xFFFFu) << 16);         // tensor_dim0[15:0] (bits 63:48)
  g1[2] = (int)(td0 >> 16);                     // tensor_dim0[31:16]; tensor_dim1=0
  g1[3] = (int)((tl0 & 0xFFFFu) << 16);         // tile_dim0 (bits 127:112)
  g1[4] = 0;                                    // tile_dim1=0 (1-D), tile_dim2=0
  g1[5] = (int)td0;                             // tensor_dim0_stride lo
  g1[6] = 0;
  g1[7] = 0;

  int4v gz = (int4v)0;
#if __clang_major__ >= 23
  __builtin_amdgcn_tensor_load_to_lds(g0, g1, gz, gz, (int8v)0, 0);
#else
  __builtin_amdgcn_tensor_load_to_lds(g0, g1, gz, gz, 0);
#endif
}
#endif

// ---------- elementwise f32 -> f16 ----------

__global__ void f32_to_f16_kernel(const float* __restrict__ in,
                                  _Float16* __restrict__ out, int n) {
  int gid = blockIdx.x * blockDim.x + threadIdx.x;
  if (gid < n) out[gid] = (_Float16)in[gid];
}

// ---------- WMMA GEMM: C[M,N] = A[M,K](f16) @ B[K,N](f16) + bias ----------
// block = 256 threads = 8 waves; block tile 64(M) x 128(N); k-step 32.

template <bool OUT_F32>
__global__ __launch_bounds__(256) void gemm_wmma_kernel(
    const _Float16* __restrict__ A, const _Float16* __restrict__ B,
    const float* __restrict__ bias, void* __restrict__ C, int M, int N, int K) {
  __shared__ _Float16 Asub[64 * 32];   // [m][k], stride 32
  __shared__ _Float16 Bt[128 * 32];    // [n][k], stride 32 (transposed)

  const int tid  = threadIdx.x;
  const int lane = tid & 31;
  const int wave = tid >> 5;
  const int wm = wave & 3;
  const int wn = wave >> 2;
  const int mbase = blockIdx.y * 64;
  const int nbase = blockIdx.x * 128;

  float8 acc[4];
#pragma unroll
  for (int i = 0; i < 4; ++i) acc[i] = (float8)(0.0f);

  const int a_m  = (tid * 8) >> 5;
  const int a_k  = (tid * 8) & 31;
  const int b_kk = tid >> 3;
  const int b_nn = (tid & 7) * 16;

  for (int k0 = 0; k0 < K; k0 += 32) {
    __syncthreads();
    *reinterpret_cast<half8*>(&Asub[a_m * 32 + a_k]) =
        *reinterpret_cast<const half8*>(&A[(size_t)(mbase + a_m) * K + k0 + a_k]);
    {
      const _Float16* src = &B[(size_t)(k0 + b_kk) * N + nbase + b_nn];
      half8 b0 = *reinterpret_cast<const half8*>(src);
      half8 b1 = *reinterpret_cast<const half8*>(src + 8);
#pragma unroll
      for (int i = 0; i < 8; ++i) {
        Bt[(b_nn + i) * 32 + b_kk]     = b0[i];
        Bt[(b_nn + 8 + i) * 32 + b_kk] = b1[i];
      }
    }
    if (k0 + 32 < K) {  // -> global_prefetch_b8
      __builtin_prefetch(&A[(size_t)(mbase + a_m) * K + k0 + 32 + a_k], 0, 1);
      __builtin_prefetch(&B[(size_t)(k0 + 32 + b_kk) * N + nbase + b_nn], 0, 1);
    }
    __syncthreads();

    const int arow = wm * 16 + (lane & 15);
    const int klo  = (lane & 16) ? 8 : 0;
    half16 af = ld2x8(&Asub[arow * 32 + klo], &Asub[arow * 32 + 16 + klo]);

    const int kb = (lane & 16) ? 16 : 0;
    // load all four B fragments first so the scheduler can clause the DS loads
    half16 bf[4];
#pragma unroll
    for (int nb = 0; nb < 4; ++nb) {
      const int col = wn * 64 + nb * 16 + (lane & 15);
      bf[nb] = ld2x8(&Bt[col * 32 + klo], &Bt[col * 32 + 16 + klo]);
    }
#pragma unroll
    for (int nb = 0; nb < 4; ++nb) acc[nb] = wmma_f16(af, bf[nb], acc[nb]);
  }

  const int row0 = mbase + wm * 16 + ((lane & 16) ? 8 : 0);
  const int col0 = nbase + wn * 64 + (lane & 15);
#pragma unroll
  for (int nb = 0; nb < 4; ++nb) {
    const int col = col0 + nb * 16;
    const float bs = bias[col];
#pragma unroll
    for (int i = 0; i < 8; ++i) {
      float v = acc[nb][i] + bs;
      if constexpr (OUT_F32)
        ((float*)C)[(size_t)(row0 + i) * N + col] = v;
      else
        ((_Float16*)C)[(size_t)(row0 + i) * N + col] = (_Float16)v;
    }
  }
}

// ---------- LSH buckets ----------

__global__ void lsh_bucket_kernel(const _Float16* __restrict__ qk16,
                                  const float* __restrict__ rot,
                                  int* __restrict__ buckets) {
  int gid = blockIdx.x * blockDim.x + threadIdx.x;
  if (gid >= BHL_) return;
  int l  = gid & (L_ - 1);
  int bh = gid / L_;
  int h  = bh & (H_ - 1);
  int b  = bh / H_;
  const _Float16* qp = qk16 + ((size_t)(b * L_ + l)) * (2 * D_) + h * HD_;
  float d[4] = {0.f, 0.f, 0.f, 0.f};
#pragma unroll 8
  for (int k = 0; k < HD_; ++k) {
    float qv = (float)qp[k];
#pragma unroll
    for (int r = 0; r < 4; ++r) d[r] += qv * rot[(h * 4 + r) * HD_ + k];
  }
  int best = 0;
  float bv = d[0];
#pragma unroll
  for (int i = 1; i < 8; ++i) {
    float c = (i < 4) ? d[i] : -d[i - 4];
    if (c > bv) { bv = c; best = i; }
  }
  buckets[gid] = best;
}

// ---------- stable counting sort over 8 buckets ----------

__global__ __launch_bounds__(256) void sort_kernel(const int* __restrict__ buckets,
                                                   int* __restrict__ idx,
                                                   int* __restrict__ rst) {
  __shared__ int cnt[8 * 256];
  const int t  = threadIdx.x;
  const int bh = blockIdx.x;
  const int* bk = buckets + (size_t)bh * L_;
  int lb[8];
  int lc[8] = {0, 0, 0, 0, 0, 0, 0, 0};
#pragma unroll
  for (int e = 0; e < 8; ++e) { lb[e] = bk[t * 8 + e]; lc[lb[e]]++; }
#pragma unroll
  for (int b = 0; b < 8; ++b) cnt[b * 256 + t] = lc[b];
  __syncthreads();
  if (t == 0) {
    int run = 0;
    for (int i = 0; i < 8 * 256; ++i) { int c = cnt[i]; cnt[i] = run; run += c; }
  }
  __syncthreads();
  int off[8];
#pragma unroll
  for (int b = 0; b < 8; ++b) off[b] = cnt[b * 256 + t];
#pragma unroll
  for (int e = 0; e < 8; ++e) {
    int l = t * 8 + e;
    int pos = off[lb[e]]++;
    idx[(size_t)bh * L_ + pos] = l;
    rst[(size_t)bh * L_ + l]   = pos;
  }
}

// ---------- gather sorted q/k/v into head-major [B,H,L,hd] f16 ----------

__global__ void gather_kernel(const _Float16* __restrict__ qk16,
                              const _Float16* __restrict__ v16,
                              const int* __restrict__ idx,
                              _Float16* __restrict__ sq, _Float16* __restrict__ sk,
                              _Float16* __restrict__ sv) {
  int gid = blockIdx.x * blockDim.x + threadIdx.x;
  if (gid >= BHL_) return;
  int bh = gid / L_;
  int h  = bh & (H_ - 1);
  int b  = bh / H_;
  int l  = idx[gid];
  const _Float16* qp = qk16 + ((size_t)(b * L_ + l)) * (2 * D_) + h * HD_;
  const _Float16* kp = qp + D_;
  const _Float16* vp = v16 + ((size_t)(b * L_ + l)) * D_ + h * HD_;
  _Float16* dq = sq + (size_t)gid * HD_;
  _Float16* dk = sk + (size_t)gid * HD_;
  _Float16* dv = sv + (size_t)gid * HD_;
#pragma unroll
  for (int c = 0; c < HD_ / 8; ++c) {
    reinterpret_cast<half8*>(dq)[c] = reinterpret_cast<const half8*>(qp)[c];
    reinterpret_cast<half8*>(dk)[c] = reinterpret_cast<const half8*>(kp)[c];
    reinterpret_cast<half8*>(dv)[c] = reinterpret_cast<const half8*>(vp)[c];
  }
}

// ---------- flash attention over sorted sequence, WMMA f16 + TDM K-tile ----------
// grid (L/64, B*H); block = 128 (4 waves); wave owns 16 query rows.

__global__ __launch_bounds__(128) void lsh_attn_kernel(
    const _Float16* __restrict__ SQ, const _Float16* __restrict__ SK,
    const _Float16* __restrict__ SV, _Float16* __restrict__ O) {
  __shared__ _Float16 Kt[32 * HD_];        // [key][hd] K tile (TDM-staged)
  __shared__ _Float16 Vt[HD_ * 32];        // [hd][key] transposed V tile
  __shared__ _Float16 Pl[4 * 16 * 32];     // per-wave P buffer

  const int tid  = threadIdx.x;
  const int lane = tid & 31;
  const int wave = tid >> 5;
  const int bh   = blockIdx.y;
  const int r0   = blockIdx.x * 64 + wave * 16;

  const _Float16* q = SQ + (size_t)bh * L_ * HD_;
  const _Float16* k = SK + (size_t)bh * L_ * HD_;
  const _Float16* v = SV + (size_t)bh * L_ * HD_;

  const int klo = (lane & 16) ? 8 : 0;
  const int kb  = (lane & 16) ? 16 : 0;
  const _Float16* qp = q + (size_t)(r0 + (lane & 15)) * HD_;
  half16 aq0 = ld2x8(qp + klo, qp + 16 + klo);
  half16 aq1 = ld2x8(qp + 32 + klo, qp + 48 + klo);

  float8 m, l, acc[4];
#pragma unroll
  for (int i = 0; i < 8; ++i) { m[i] = -1e30f; l[i] = 0.f; }
#pragma unroll
  for (int nb = 0; nb < 4; ++nb) acc[nb] = (float8)(0.0f);

  _Float16* pb = &Pl[wave * 16 * 32];
  const float scale = 0.125f;  // 1/sqrt(64)

  const int v_key = (tid * 16) >> 6;   // 0..31
  const int v_h0  = (tid * 16) & 63;   // 0,16,32,48

#pragma unroll 1
  for (int j = 0; j < L_; j += 32) {
    __syncthreads();
#if HAVE_TDM
    if (wave == 0) {  // stage 32x64 K tile (4KB contiguous) via Tensor Data Mover
      tdm_load_tile_4k(k + (size_t)j * HD_, Kt);
      __builtin_amdgcn_s_wait_tensorcnt((short)0);
    }
#else
    {  // fallback: cooperative copy of K tile
      _Float16* dst = &Kt[tid * 16];
      const _Float16* src = k + (size_t)j * HD_ + tid * 16;
      reinterpret_cast<half8*>(dst)[0] = reinterpret_cast<const half8*>(src)[0];
      reinterpret_cast<half8*>(dst)[1] = reinterpret_cast<const half8*>(src)[1];
    }
#endif
    {  // load V tile (32 keys x 64 hd) transposed into Vt[hd][key]
      const _Float16* src = v + (size_t)(j + v_key) * HD_ + v_h0;
      half8 a0 = *reinterpret_cast<const half8*>(src);
      half8 a1 = *reinterpret_cast<const half8*>(src + 8);
#pragma unroll
      for (int i = 0; i < 8; ++i) {
        Vt[(v_h0 + i) * 32 + v_key]     = a0[i];
        Vt[(v_h0 + 8 + i) * 32 + v_key] = a1[i];
      }
    }
    __syncthreads();

    // S(16x32) = Q @ K^T : load all four K B-fragments, then issue WMMAs
    const _Float16* kp0 = &Kt[(lane & 15) * HD_];
    const _Float16* kp1 = kp0 + 16 * HD_;
    half16 b00 = ld2x8(kp0 + klo, kp0 + 16 + klo);
    half16 b01 = ld2x8(kp0 + 32 + klo, kp0 + 48 + klo);
    half16 b10 = ld2x8(kp1 + klo, kp1 + 16 + klo);
    half16 b11 = ld2x8(kp1 + 32 + klo, kp1 + 48 + klo);
    float8 s0 = (float8)(0.0f), s1 = (float8)(0.0f);
    s0 = wmma_f16(aq0, b00, s0);
    s1 = wmma_f16(aq0, b10, s1);
    s0 = wmma_f16(aq1, b01, s0);
    s1 = wmma_f16(aq1, b11, s1);

    // online softmax
    float8 t;
#pragma unroll
    for (int i = 0; i < 8; ++i) {
      s0[i] *= scale; s1[i] *= scale;
      t[i] = fmaxf(s0[i], s1[i]);
    }
#pragma unroll
    for (int msk = 1; msk < 16; msk <<= 1)
#pragma unroll
      for (int i = 0; i < 8; ++i) t[i] = fmaxf(t[i], __shfl_xor(t[i], msk, 32));

    float8 mn, al, rs;
#pragma unroll
    for (int i = 0; i < 8; ++i) {
      mn[i] = fmaxf(m[i], t[i]);
      al[i] = __expf(m[i] - mn[i]);
      s0[i] = __expf(s0[i] - mn[i]);
      s1[i] = __expf(s1[i] - mn[i]);
      rs[i] = s0[i] + s1[i];
    }
#pragma unroll
    for (int msk = 1; msk < 16; msk <<= 1)
#pragma unroll
      for (int i = 0; i < 8; ++i) rs[i] += __shfl_xor(rs[i], msk, 32);
#pragma unroll
    for (int i = 0; i < 8; ++i) { l[i] = l[i] * al[i] + rs[i]; m[i] = mn[i]; }
#pragma unroll
    for (int nb = 0; nb < 4; ++nb)
#pragma unroll
      for (int i = 0; i < 8; ++i) acc[nb][i] *= al[i];

    // C-layout -> A-layout reshape of P through per-wave LDS
    {
      const int rb = (lane & 16) ? 8 : 0;
      const int n  = lane & 15;
#pragma unroll
      for (int i = 0; i < 8; ++i) {
        pb[(rb + i) * 32 + n]      = (_Float16)s0[i];
        pb[(rb + i) * 32 + 16 + n] = (_Float16)s1[i];
      }
    }
    asm volatile("s_wait_dscnt 0" ::: "memory");  // intra-wave DS ordering

    const int prow = lane & 15;
    half16 ap = ld2x8(&pb[prow * 32 + klo], &pb[prow * 32 + 16 + klo]);

    // load all four V B-fragments, then issue WMMAs
    half16 bvf[4];
#pragma unroll
    for (int nb = 0; nb < 4; ++nb) {
      const int hdc = nb * 16 + (lane & 15);
      bvf[nb] = ld2x8(&Vt[hdc * 32 + klo], &Vt[hdc * 32 + 16 + klo]);
    }
#pragma unroll
    for (int nb = 0; nb < 4; ++nb) acc[nb] = wmma_f16(ap, bvf[nb], acc[nb]);
  }

  const int orow = r0 + ((lane & 16) ? 8 : 0);
#pragma unroll
  for (int nb = 0; nb < 4; ++nb)
#pragma unroll
    for (int i = 0; i < 8; ++i) {
      float o = acc[nb][i] / l[i];
      O[(size_t)(bh * L_ + orow + i) * HD_ + nb * 16 + (lane & 15)] = (_Float16)o;
    }
}

// ---------- un-sort + merge heads into [B*L, D] f16 ----------

__global__ void unsort_kernel(const _Float16* __restrict__ osort,
                              const int* __restrict__ rst,
                              _Float16* __restrict__ merged) {
  int gid = blockIdx.x * blockDim.x + threadIdx.x;
  if (gid >= BHL_) return;
  int l  = gid & (L_ - 1);
  int bh = gid / L_;
  int h  = bh & (H_ - 1);
  int b  = bh / H_;
  int pos = rst[gid];
  const _Float16* src = osort + ((size_t)bh * L_ + pos) * HD_;
  _Float16* dst = merged + ((size_t)(b * L_ + l)) * D_ + h * HD_;
#pragma unroll
  for (int c = 0; c < HD_ / 8; ++c)
    reinterpret_cast<half8*>(dst)[c] = reinterpret_cast<const half8*>(src)[c];
}

// ---------- launcher ----------

extern "C" void kernel_launch(void* const* d_in, const int* in_sizes, int n_in,
                              void* d_out, int out_size, void* d_ws, size_t ws_size,
                              hipStream_t stream) {
  const float* x   = (const float*)d_in[0];
  const float* Wqk = (const float*)d_in[1];
  const float* bqk = (const float*)d_in[2];
  const float* Wv  = (const float*)d_in[3];
  const float* bv  = (const float*)d_in[4];
  const float* Wo  = (const float*)d_in[5];
  const float* bo  = (const float*)d_in[6];
  const float* rot = (const float*)d_in[7];

  char* base = (char*)d_ws;
  size_t off = 0;
  auto take = [&](size_t bytes) -> void* {
    void* r = base + off;
    off += (bytes + 255) & ~(size_t)255;
    return r;
  };

  const size_t BL = (size_t)B_ * L_;  // 8192
  _Float16* x16    = (_Float16*)take(BL * D_ * 2);
  _Float16* wqk16  = (_Float16*)take((size_t)D_ * 2 * D_ * 2);
  _Float16* wv16   = (_Float16*)take((size_t)D_ * D_ * 2);
  _Float16* wo16   = (_Float16*)take((size_t)D_ * D_ * 2);
  _Float16* qk16   = (_Float16*)take(BL * 2 * D_ * 2);
  _Float16* v16b   = (_Float16*)take(BL * D_ * 2);
  int*      bkt    = (int*)take((size_t)BHL_ * 4);
  int*      idx    = (int*)take((size_t)BHL_ * 4);
  int*      rst    = (int*)take((size_t)BHL_ * 4);
  _Float16* sq     = (_Float16*)take((size_t)BHL_ * HD_ * 2);
  _Float16* sk     = (_Float16*)take((size_t)BHL_ * HD_ * 2);
  _Float16* sv     = (_Float16*)take((size_t)BHL_ * HD_ * 2);
  _Float16* osort  = (_Float16*)take((size_t)BHL_ * HD_ * 2);
  _Float16* merged = (_Float16*)take(BL * D_ * 2);

  auto cvt = [&](const float* in, _Float16* out, int n) {
    f32_to_f16_kernel<<<(n + 255) / 256, 256, 0, stream>>>(in, out, n);
  };
  cvt(x, x16, (int)(BL * D_));
  cvt(Wqk, wqk16, D_ * 2 * D_);
  cvt(Wv, wv16, D_ * D_);
  cvt(Wo, wo16, D_ * D_);

  gemm_wmma_kernel<false><<<dim3((2 * D_) / 128, BL / 64), 256, 0, stream>>>(
      x16, wqk16, bqk, qk16, (int)BL, 2 * D_, D_);
  gemm_wmma_kernel<false><<<dim3(D_ / 128, BL / 64), 256, 0, stream>>>(
      x16, wv16, bv, v16b, (int)BL, D_, D_);

  lsh_bucket_kernel<<<(BHL_ + 255) / 256, 256, 0, stream>>>(qk16, rot, bkt);
  sort_kernel<<<B_ * H_, 256, 0, stream>>>(bkt, idx, rst);
  gather_kernel<<<BHL_ / 256, 256, 0, stream>>>(qk16, v16b, idx, sq, sk, sv);

  lsh_attn_kernel<<<dim3(L_ / 64, B_ * H_), 128, 0, stream>>>(sq, sk, sv, osort);

  unsort_kernel<<<BHL_ / 256, 256, 0, stream>>>(osort, rst, merged);

  gemm_wmma_kernel<true><<<dim3(D_ / 128, BL / 64), 256, 0, stream>>>(
      merged, wo16, bo, (float*)d_out, (int)BL, D_, D_);
}
